// Encoder_11751030522482
// MI455X (gfx1250) — matrix operand order repaired and checked by reading, not compiled, not run on hardware
//
#include <hip/hip_runtime.h>
#include <hip/hip_bf16.h>

typedef __attribute__((ext_vector_type(16))) __bf16 v16bf;
typedef __attribute__((ext_vector_type(8)))  __bf16 v8bf;
typedef __attribute__((ext_vector_type(8)))  float  v8f;

#define B_   256
#define T_   128
#define WD_  400
#define KX_  416   // WD padded to multiple of 32
#define H_   500
#define KH_  512   // H padded to multiple of 32
#define FPD_ 60
#define KZ_  64    // FPD padded
#define NG_  2000  // 4H
#define NZ_  1000  // 2H
#define NZP_ 1008  // 2H padded to multiple of 16

__device__ __forceinline__ float sigmoidf_(float x) { return 1.0f / (1.0f + __expf(-x)); }

// 16-bit A-matrix 16x32 fragment (row-major source, ld = K):
// lanes 0-15 hold row M=lane, K = {k0..k0+7, k0+16..k0+23}
// lanes 16-31 hold row M=lane-16, K = {k0+8..k0+15, k0+24..k0+31}
__device__ __forceinline__ v16bf load_a_frag(const __bf16* A, int lda, int row0, int k0, int lane) {
    int lh = (lane >> 4) & 1;
    int r  = lane & 15;
    const __bf16* p = A + (size_t)(row0 + r) * lda + k0 + lh * 8;
    v8bf lo = *(const v8bf*)p;
    v8bf hi = *(const v8bf*)(p + 16);
    v16bf f;
#pragma unroll
    for (int i = 0; i < 8; ++i) { f[i] = lo[i]; f[i + 8] = hi[i]; }
    return f;
}

// 16-bit B-matrix 32x16 fragment from transposed weights Bt[N][K] (ld = K):
// lanes 0-15 hold col N=lane, K = {k0..k0+7, k0+16..k0+23}; lanes 16-31 hold K = {k0+8..k0+15, k0+24..k0+31}
__device__ __forceinline__ v16bf load_b_frag(const __bf16* Bt, int ldb, int col0, int k0, int lane) {
    int lh = (lane >> 4) & 1;
    int c  = lane & 15;
    const __bf16* p = Bt + (size_t)(col0 + c) * ldb + k0 + lh * 8;
    v8bf lo = *(const v8bf*)p;
    v8bf hi = *(const v8bf*)(p + 16);
    v16bf f;
#pragma unroll
    for (int i = 0; i < 8; ++i) { f[i] = lo[i]; f[i + 8] = hi[i]; }
    return f;
}

// ---------- weight transpose + bf16 pack (padded) ----------
__global__ void prep_weights(const float* __restrict__ W, const float* __restrict__ Wz,
                             __bf16* __restrict__ Wxt, __bf16* __restrict__ Wht,
                             __bf16* __restrict__ Wzt) {
    const int nWx = NG_ * KX_;
    const int nWh = NG_ * KH_;
    const int nWz = NZP_ * KZ_;
    const int total = nWx + nWh + nWz;
    for (int i = blockIdx.x * blockDim.x + threadIdx.x; i < total; i += gridDim.x * blockDim.x) {
        if (i < nWx) {
            int n = i / KX_, k = i - n * KX_;
            Wxt[i] = (k < WD_) ? (__bf16)W[(size_t)k * NG_ + n] : (__bf16)0.0f;
        } else if (i < nWx + nWh) {
            int j = i - nWx;
            int n = j / KH_, k = j - n * KH_;
            Wht[j] = (k < H_) ? (__bf16)W[(size_t)(WD_ + k) * NG_ + n] : (__bf16)0.0f;
        } else {
            int j = i - nWx - nWh;
            int n = j / KZ_, k = j - n * KZ_;
            Wzt[j] = (n < NZ_ && k < FPD_) ? (__bf16)Wz[(size_t)k * NZ_ + n] : (__bf16)0.0f;
        }
    }
}

// ---------- embedding gather ----------
__global__ void gather_embed(const int* __restrict__ words, const int* __restrict__ fields,
                             const int* __restrict__ pos, const int* __restrict__ rpos,
                             const float* __restrict__ Ew, const float* __restrict__ Ef,
                             const float* __restrict__ Ep, const float* __restrict__ Erp,
                             __bf16* __restrict__ xw, __bf16* __restrict__ z,
                             float* __restrict__ fpe) {
    int row = blockIdx.x;                       // row = b*T + t
    int w = words[row], f = fields[row], p = pos[row], rp = rpos[row];
    const float* er = Ew + (size_t)w * WD_;
    __bf16* xr = xw + (size_t)row * KX_;
    for (int j = threadIdx.x; j < KX_; j += blockDim.x)
        xr[j] = (j < WD_) ? (__bf16)er[j] : (__bf16)0.0f;
    __bf16* zr = z + (size_t)row * KZ_;
    float*  fr = fpe + (size_t)row * FPD_;
    for (int j = threadIdx.x; j < KZ_; j += blockDim.x) {
        float v = 0.0f;
        if (j < 50)      v = Ef[(size_t)f * 50 + j];
        else if (j < 55) v = Ep[p * 5 + (j - 50)];
        else if (j < 60) v = Erp[rp * 5 + (j - 55)];
        zr[j] = (__bf16)v;
        if (j < FPD_) fr[j] = v;
    }
}

__global__ void init_state(__bf16* __restrict__ h, float* __restrict__ c) {
    int i = blockIdx.x * blockDim.x + threadIdx.x;
    if (i < B_ * KH_) h[i] = (__bf16)0.0f;
    if (i < B_ * H_)  c[i] = 0.0f;
}

// ---------- bf16 WMMA GEMM, register-blocked 2 M-tiles x 4 N-tiles per wave ----------
// EPI 0: + bias, store bf16 (gates_x).  EPI 1: + bias, sigmoid for n<H, tanh for n>=H (lzact).
template <int EPI>
__global__ void gemm_bf16_wmma(const __bf16* __restrict__ A, const __bf16* __restrict__ Bt,
                               const float* __restrict__ bias, __bf16* __restrict__ Cb,
                               int M, int Ntiles, int Nstore, int K, int ldc) {
    int gw   = blockIdx.x * (blockDim.x >> 5) + (threadIdx.x >> 5);
    int lane = threadIdx.x & 31;
    int mg = M >> 5;                 // groups of 2 m-tiles (32 rows)
    int ng = (Ntiles + 3) >> 2;      // groups of 4 n-tiles (64 cols)
    if (gw >= mg * ng) return;
    int gm = gw / ng, gn = gw - gm * ng;
    int row0 = gm << 5;
    int tn0  = gn << 2;
    v8f acc[2][4] = {};
    for (int k0 = 0; k0 < K; k0 += 32) {
        if (k0 + 32 < K)   // lowers to global_prefetch_b8
            __builtin_prefetch(A + (size_t)(row0 + (lane & 15)) * K + k0 + 32, 0, 1);
        v16bf a0 = load_a_frag(A, K, row0,      k0, lane);
        v16bf a1 = load_a_frag(A, K, row0 + 16, k0, lane);
#pragma unroll
        for (int ni = 0; ni < 4; ++ni) {
            int tn = tn0 + ni; if (tn >= Ntiles) tn = Ntiles - 1;   // clamp (stores guarded)
            v16bf b = load_b_frag(Bt, K, tn << 4, k0, lane);
            acc[0][ni] = __builtin_amdgcn_wmma_f32_16x16x32_bf16(false, a0, false, b, (short)0, acc[0][ni], false, false);
            acc[1][ni] = __builtin_amdgcn_wmma_f32_16x16x32_bf16(false, a1, false, b, (short)0, acc[1][ni], false, false);
        }
    }
    int lh = (lane >> 4) & 1, cl = lane & 15;
#pragma unroll
    for (int ni = 0; ni < 4; ++ni) {
        int tn = tn0 + ni;
        if (tn >= Ntiles) break;
        int col = (tn << 4) + cl;
        if (col >= Nstore) continue;
        float bv = bias[col];
#pragma unroll
        for (int mi = 0; mi < 2; ++mi) {
#pragma unroll
            for (int j = 0; j < 8; ++j) {
                int r = row0 + mi * 16 + j + lh * 8;
                float v = acc[mi][ni][j] + bv;
                if (EPI == 1) v = (col < H_) ? sigmoidf_(v) : tanhf(v);
                Cb[(size_t)r * ldc + col] = (__bf16)v;
            }
        }
    }
}

// ---------- one recurrent timestep ----------
// grid = B/16 blocks, 256 threads (8 waves). Each block owns 16 batch rows; h/c updated in place.
// __launch_bounds__(256, 1): latency-bound kernel at deliberate single-occupancy -> give the
// allocator the full register file so the 16 hoisted A fragments (128 VGPRs) never spill.
__global__ void __launch_bounds__(256, 1)
fglstm_step(const __bf16* __restrict__ gx, const __bf16* __restrict__ lz,
            const __bf16* __restrict__ Wht, __bf16* __restrict__ h,
            float* __restrict__ c, float* __restrict__ outs,
            float* __restrict__ h_f, float* __restrict__ c_f, int t) {
    extern __shared__ char smem[];
    float*  gl = (float*)smem;                                       // [16][NG_]   (128 KB)
    __bf16* aT = (__bf16*)(smem + (size_t)16 * NG_ * sizeof(float)); // [16][KH_]   (16 KB)
    int wave = threadIdx.x >> 5, lane = threadIdx.x & 31;
    int row0 = blockIdx.x << 4;
    int lh = (lane >> 4) & 1, cl = lane & 15;

    // Phase 0: async global->LDS copy of this block's h tile (rows are contiguous: 16 KB)
    {
        const char* gsrc = (const char*)(h + (size_t)row0 * KH_);
#pragma unroll
        for (int it = 0; it < 4; ++it) {
            unsigned off = threadIdx.x * 16u + (unsigned)it * 4096u;
            unsigned ldsAddr = (unsigned)(uintptr_t)aT + off;            // low 32 bits = LDS offset
            unsigned long long ga = (unsigned long long)(uintptr_t)(gsrc + off);
            asm volatile("global_load_async_to_lds_b128 %0, %1, off"
                         :: "v"(ldsAddr), "v"(ga) : "memory");
        }
        asm volatile("s_wait_asynccnt 0x0" ::: "memory");
    }
    __syncthreads();

    // Hoist all 16 A fragments (invariant across n-tiles) from LDS into registers
    v16bf afr[KH_ / 32];
#pragma unroll
    for (int kc = 0; kc < KH_ / 32; ++kc) {
        const __bf16* p = aT + cl * KH_ + kc * 32 + lh * 8;
        v8bf lo = *(const v8bf*)p;
        v8bf hi = *(const v8bf*)(p + 16);
#pragma unroll
        for (int i = 0; i < 8; ++i) { afr[kc][i] = lo[i]; afr[kc][8 + i] = hi[i]; }
    }

    // Phase 1: gates = h @ Wh^T + gates_x -> LDS
    for (int tn = wave; tn < NG_ / 16; tn += (blockDim.x >> 5)) {
        int col0 = tn << 4;
        v8f acc = {};
#pragma unroll
        for (int kc = 0; kc < KH_ / 32; ++kc) {
            v16bf b = load_b_frag(Wht, KH_, col0, kc * 32, lane);
            acc = __builtin_amdgcn_wmma_f32_16x16x32_bf16(false, afr[kc], false, b, (short)0, acc, false, false);
        }
        int col = col0 + cl;
#pragma unroll
        for (int j = 0; j < 8; ++j) {
            int r = j + lh * 8;
            size_t grow = (size_t)(row0 + r) * T_ + t;
            gl[r * NG_ + col] = acc[j] + (float)gx[grow * NG_ + col];
        }
    }
    __syncthreads();

    // Phase 2: c = f*c + i*g + l*zh ; h = o*tanh(c)
    for (int idx = threadIdx.x; idx < 16 * H_; idx += blockDim.x) {
        int r = idx / H_, n = idx - r * H_;
        int b = row0 + r;
        size_t grow = (size_t)b * T_ + t;
        float gi = sigmoidf_(gl[r * NG_ + n]);
        float gf = sigmoidf_(gl[r * NG_ + H_ + n]);
        float go = sigmoidf_(gl[r * NG_ + 2 * H_ + n]);
        float gg = tanhf    (gl[r * NG_ + 3 * H_ + n]);
        float lv = (float)lz[grow * NZ_ + n];
        float zh = (float)lz[grow * NZ_ + H_ + n];
        float cc = gf * c[(size_t)b * H_ + n] + gi * gg + lv * zh;
        float hh = go * tanhf(cc);
        c[(size_t)b * H_ + n] = cc;
        h[(size_t)b * KH_ + n] = (__bf16)hh;      // cols H_..KH_-1 stay zero from init
        outs[grow * H_ + n] = hh;
        h_f[(size_t)b * H_ + n] = hh;             // last step's write survives
        c_f[(size_t)b * H_ + n] = cc;
    }
}

extern "C" void kernel_launch(void* const* d_in, const int* in_sizes, int n_in,
                              void* d_out, int out_size, void* d_ws, size_t ws_size,
                              hipStream_t stream) {
    (void)in_sizes; (void)n_in; (void)out_size; (void)ws_size;
    const int*   words  = (const int*)d_in[0];
    const int*   fields = (const int*)d_in[1];
    const int*   pos    = (const int*)d_in[2];
    const int*   rpos   = (const int*)d_in[3];
    const float* Ew     = (const float*)d_in[4];
    const float* Ef     = (const float*)d_in[5];
    const float* Ep     = (const float*)d_in[6];
    const float* Erp    = (const float*)d_in[7];
    const float* W      = (const float*)d_in[8];
    const float* bias   = (const float*)d_in[9];
    const float* Wz     = (const float*)d_in[10];
    const float* bz     = (const float*)d_in[11];

    float* out  = (float*)d_out;
    float* outs = out;                                   // [B,T,H]
    float* h_f  = out + (size_t)B_ * T_ * H_;            // [B,H]
    float* c_f  = h_f + (size_t)B_ * H_;                 // [B,H]
    float* fpe  = c_f + (size_t)B_ * H_;                 // [B,T,FPD]

    char* wsp = (char*)d_ws;
    auto carve = [&](size_t bytes) { char* p = wsp; wsp += (bytes + 255) & ~(size_t)255; return p; };
    __bf16* xw  = (__bf16*)carve((size_t)B_ * T_ * KX_ * 2);
    __bf16* z   = (__bf16*)carve((size_t)B_ * T_ * KZ_ * 2);
    __bf16* Wxt = (__bf16*)carve((size_t)NG_ * KX_ * 2);
    __bf16* Wht = (__bf16*)carve((size_t)NG_ * KH_ * 2);
    __bf16* Wzt = (__bf16*)carve((size_t)NZP_ * KZ_ * 2);
    __bf16* gx  = (__bf16*)carve((size_t)B_ * T_ * NG_ * 2);
    __bf16* lzA = (__bf16*)carve((size_t)B_ * T_ * NZ_ * 2);
    __bf16* h   = (__bf16*)carve((size_t)B_ * KH_ * 2);
    float*  c   = (float*) carve((size_t)B_ * H_ * 4);

    prep_weights<<<1024, 256, 0, stream>>>(W, Wz, Wxt, Wht, Wzt);
    gather_embed<<<B_ * T_, 128, 0, stream>>>(words, fields, pos, rpos, Ew, Ef, Ep, Erp, xw, z, fpe);
    init_state<<<(B_ * KH_ + 255) / 256, 256, 0, stream>>>(h, c);

    {   // gates_x = x @ Wx + b : M=32768, N=2000, K=416 ; 2x4 tile groups
        int groups = ((B_ * T_) >> 5) * (((NG_ / 16) + 3) >> 2);
        gemm_bf16_wmma<0><<<(groups + 7) / 8, 256, 0, stream>>>(xw, Wxt, bias, gx,
                                                                B_ * T_, NG_ / 16, NG_, KX_, NG_);
    }
    {   // lzact = act(z @ Wz + bz) : M=32768, N=1000 (63 padded tiles), K=64
        int groups = ((B_ * T_) >> 5) * (((NZP_ / 16) + 3) >> 2);
        gemm_bf16_wmma<1><<<(groups + 7) / 8, 256, 0, stream>>>(z, Wzt, bz, lzA,
                                                                B_ * T_, NZP_ / 16, NZ_, KZ_, NZ_);
    }
    size_t shmem = (size_t)16 * NG_ * sizeof(float) + (size_t)16 * KH_ * sizeof(__bf16);
    for (int t = 0; t < T_; ++t)
        fglstm_step<<<B_ / 16, 256, shmem, stream>>>(gx, lzA, Wht, h, c, outs, h_f, c_f, t);
}
